// convex_hull_cvx_dyn_1786706395171
// MI455X (gfx1250) — matrix operand order repaired and checked by reading, not compiled, not run on hardware
//
#include <hip/hip_runtime.h>
#include <hip/hip_bf16.h>
#include <cmath>
#include <cstddef>
#include <cstdint>

// ---------------------------------------------------------------------------
// Types for CDNA5 WMMA (wave32): v_wmma_f32_16x16x32_f16
// ---------------------------------------------------------------------------
typedef __attribute__((ext_vector_type(16))) _Float16 v16h;
typedef __attribute__((ext_vector_type(8)))  _Float16 v8h;
typedef __attribute__((ext_vector_type(8)))  float    v8f;

#define WAVE 32

// ---------------------------------------------------------------------------
// CDNA5 async global->LDS copy (ASYNCcnt-tracked), per CDNA5 ISA
// ---------------------------------------------------------------------------
__device__ __forceinline__ void async_copy_b128(unsigned lds_addr, const _Float16* g) {
  asm volatile("global_load_async_to_lds_b128 %0, %1, off"
               :: "v"(lds_addr), "v"(g) : "memory");
}
__device__ __forceinline__ void wait_async0() {
  asm volatile("s_wait_asynccnt 0" ::: "memory");
}

// ---------------------------------------------------------------------------
// Row normalization: out = row / ||row||, emit f32 + f16 copies.
// ---------------------------------------------------------------------------
__global__ __launch_bounds__(256)
void k_normalize_rows(const float* __restrict__ in, float* __restrict__ outf,
                      _Float16* __restrict__ outh, int D) {
  const int row  = blockIdx.x;
  const int lane = threadIdx.x & 31;
  const int wv   = threadIdx.x >> 5;
  const float* src = in + (size_t)row * D;

  float s = 0.f;
  for (int c = threadIdx.x; c < D; c += blockDim.x) { float v = src[c]; s += v * v; }
  #pragma unroll
  for (int off = 16; off; off >>= 1) s += __shfl_xor(s, off, WAVE);

  __shared__ float red[8];
  if (lane == 0) red[wv] = s;
  __syncthreads();
  if (wv == 0) {
    float t = (lane < 8) ? red[lane] : 0.f;
    #pragma unroll
    for (int off = 16; off; off >>= 1) t += __shfl_xor(t, off, WAVE);
    if (lane == 0) red[0] = t;
  }
  __syncthreads();
  const float inv = rsqrtf(red[0] + 1e-20f);
  for (int c = threadIdx.x; c < D; c += blockDim.x) {
    float v = src[c] * inv;
    outf[(size_t)row * D + c] = v;
    outh[(size_t)row * D + c] = (_Float16)v;
  }
}

// ---------------------------------------------------------------------------
// NT GEMM via WMMA f16->f32 with DOUBLE-BUFFERED LDS panel staging
// (GLOBAL_LOAD_ASYNC_TO_LDS_B128) and per-wave register blocking.
//   C[m][n] = sum_k A[m][k] * B[n][k]
// Block = 256 threads (8 waves) computes a 128x128 C macro-tile.
// Per K-step (32): stage next A/B panels (128x32 each) into the alternate
// LDS buffer while computing 8 WMMAs on the current one; then
// s_wait_asynccnt 0 + one barrier. Waves arranged 2(M) x 4(N); each wave
// computes a 64x32 sub-tile (4x2 WMMA tiles).
//   -> 16 KB global bytes per 1.05 MFLOP block step = 64 FLOP/byte,
//      with async copies overlapped against the matrix pipe.
// LDS rows padded to 40 halfs (80B stride = 20-bank stride: conflict-free,
// 16B-aligned fragment loads). 2 x (8+8) KB panels + pad = 40 KB/block.
// ---------------------------------------------------------------------------
#define LDS_STRIDE 40
#define PANEL_H   (128 * LDS_STRIDE)        // halfs per panel buffer
#define PANEL_B   (PANEL_H * 2)             // bytes per panel buffer

__global__ __launch_bounds__(256)
void k_gemm_nt_f16(const _Float16* A, const _Float16* B,
                   float* __restrict__ C, int Mrows, int Nrows, int K) {
  __shared__ _Float16 sA[2 * PANEL_H];
  __shared__ _Float16 sB[2 * PANEL_H];

  const int tid    = threadIdx.x;
  const int lane   = tid & 31;
  const int wave   = tid >> 5;
  const int tilesN = Nrows >> 7;
  const int tm     = (blockIdx.x / tilesN) << 7;
  const int tn     = (blockIdx.x % tilesN) << 7;

  const int wm = (wave >> 2) << 6;      // 0 or 64
  const int wn = (wave & 3) << 5;       // 0,32,64,96

  const int r  = lane & 15;
  const int kh = lane >> 4;

  // Staging: 512 8-half chunks per panel; thread t handles chunks {t, t+256}:
  // chunk c -> row c>>2, sub-chunk c&3.
  const int rowS = tid >> 2;
  const int subS = tid & 3;
  const unsigned ldsA0 = (unsigned)(uintptr_t)&sA[rowS * LDS_STRIDE + subS * 8];
  const unsigned ldsA1 = (unsigned)(uintptr_t)&sA[(rowS + 64) * LDS_STRIDE + subS * 8];
  const unsigned ldsB0 = (unsigned)(uintptr_t)&sB[rowS * LDS_STRIDE + subS * 8];
  const unsigned ldsB1 = (unsigned)(uintptr_t)&sB[(rowS + 64) * LDS_STRIDE + subS * 8];
  const _Float16* gA0 = A + (size_t)(tm + rowS) * K + subS * 8;
  const _Float16* gA1 = A + (size_t)(tm + rowS + 64) * K + subS * 8;
  const _Float16* gB0 = B + (size_t)(tn + rowS) * K + subS * 8;
  const _Float16* gB1 = B + (size_t)(tn + rowS + 64) * K + subS * 8;

  // Fragment base pointers in LDS (documented 16-bit WMMA layouts):
  //   A frag (16x32): lane m + 16*kh, two 8-half runs at K=8*kh and 16+8*kh.
  //   B frag (32x16): lane n + 16*kh, two 8-half runs at K=8*kh and 16+8*kh.
  const _Float16* fA = &sA[(wm + r) * LDS_STRIDE + 8 * kh];
  const _Float16* fB = &sB[(wn + r) * LDS_STRIDE + 8 * kh];

  v8f acc[4][2] = {};

  // Prologue: stage k0 = 0 into buffer 0.
  async_copy_b128(ldsA0, gA0);
  async_copy_b128(ldsA1, gA1);
  async_copy_b128(ldsB0, gB0);
  async_copy_b128(ldsB1, gB1);
  wait_async0();
  __syncthreads();

  int buf = 0;
  for (int k0 = 0; k0 < K; k0 += 32) {
    const bool more = (k0 + 32) < K;          // block-uniform
    if (more) {
      const unsigned boff = (unsigned)((buf ^ 1) * PANEL_B);
      async_copy_b128(ldsA0 + boff, gA0 + k0 + 32);
      async_copy_b128(ldsA1 + boff, gA1 + k0 + 32);
      async_copy_b128(ldsB0 + boff, gB0 + k0 + 32);
      async_copy_b128(ldsB1 + boff, gB1 + k0 + 32);
    }

    const _Float16* pa = fA + buf * PANEL_H;
    const _Float16* pb = fB + buf * PANEL_H;
    union { v16h v; v8h h[2]; } a[4], b[2];
    #pragma unroll
    for (int i = 0; i < 4; ++i) {
      const _Float16* p = pa + (size_t)(16 * i) * LDS_STRIDE;
      a[i].h[0] = *(const v8h*)(p);
      a[i].h[1] = *(const v8h*)(p + 16);
    }
    #pragma unroll
    for (int j = 0; j < 2; ++j) {
      const _Float16* p = pb + (size_t)(16 * j) * LDS_STRIDE;
      b[j].h[0] = *(const v8h*)(p);
      b[j].h[1] = *(const v8h*)(p + 16);
    }
    #pragma unroll
    for (int i = 0; i < 4; ++i)
      #pragma unroll
      for (int j = 0; j < 2; ++j)
        acc[i][j] = __builtin_amdgcn_wmma_f32_16x16x32_f16(
            false, a[i].v, false, b[j].v, (short)0, acc[i][j], false, false);

    if (more) {
      wait_async0();       // next-stage panels landed (this wave)
      __syncthreads();     // all waves: reads of buf done, writes of buf^1 done
    }
    buf ^= 1;
  }

  // C/D layout: VGPR q -> M = q + 8*kh, N = lane%16
  #pragma unroll
  for (int i = 0; i < 4; ++i)
    #pragma unroll
    for (int j = 0; j < 2; ++j) {
      float* crow = C + (size_t)(tm + wm + 16 * i + 8 * kh) * Nrows
                      + (tn + wn + 16 * j + r);
      #pragma unroll
      for (int q = 0; q < 8; ++q) crow[(size_t)q * Nrows] = acc[i][j][q];
    }
}

// ---------------------------------------------------------------------------
// f32 -> f16 elementwise convert
// ---------------------------------------------------------------------------
__global__ __launch_bounds__(256)
void k_convert_f32_f16(const float* __restrict__ in, _Float16* __restrict__ out, int n) {
  int i = blockIdx.x * blockDim.x + threadIdx.x;
  if (i < n) out[i] = (_Float16)in[i];
}

// ---------------------------------------------------------------------------
// Transpose f32 [R x Cc] -> f16 [Cc x R]
// ---------------------------------------------------------------------------
__global__ __launch_bounds__(256)
void k_transpose_f32_to_f16(const float* __restrict__ in, _Float16* __restrict__ out,
                            int R, int Cc) {
  int idx = blockIdx.x * blockDim.x + threadIdx.x;
  if (idx >= R * Cc) return;
  int rr = idx / Cc, cc = idx % Cc;
  out[(size_t)cc * R + rr] = (_Float16)in[idx];
}

// ---------------------------------------------------------------------------
// Power iteration on Gram M (1024x1024 f32), single block of 1024 threads.
// Produces step = 1 / (2*lambda_max + eps).
// ---------------------------------------------------------------------------
__global__ __launch_bounds__(1024)
void k_power_iter(const float* __restrict__ Mg, float* __restrict__ stepOut) {
  __shared__ float v[1024];
  __shared__ float red[32];
  const int tid = threadIdx.x;
  v[tid] = 1.0f / 1024.0f;
  __syncthreads();

  const float* mr = Mg + (size_t)tid * 1024;
  for (int it = 0; it < 50; ++it) {
    float w = 0.f;
    for (int j = 0; j < 1024; ++j) w += mr[j] * v[j];
    float s = w * w;
    #pragma unroll
    for (int off = 16; off; off >>= 1) s += __shfl_xor(s, off, WAVE);
    if ((tid & 31) == 0) red[tid >> 5] = s;
    __syncthreads();
    if (tid < 32) {
      float t = red[tid];
      #pragma unroll
      for (int off = 16; off; off >>= 1) t += __shfl_xor(t, off, WAVE);
      if (tid == 0) red[0] = t;
    }
    __syncthreads();
    const float inv = 1.0f / (sqrtf(red[0]) + 1e-8f);
    v[tid] = w * inv;
    __syncthreads();
  }
  float mv = 0.f;
  for (int j = 0; j < 1024; ++j) mv += mr[j] * v[j];
  float s = v[tid] * mv;
  #pragma unroll
  for (int off = 16; off; off >>= 1) s += __shfl_xor(s, off, WAVE);
  if ((tid & 31) == 0) red[tid >> 5] = s;
  __syncthreads();
  if (tid < 32) {
    float t = red[tid];
    #pragma unroll
    for (int off = 16; off; off >>= 1) t += __shfl_xor(t, off, WAVE);
    if (tid == 0) *stepOut = 1.0f / (2.0f * t + 1e-8f);
  }
}

// ---------------------------------------------------------------------------
// FISTA state init: X = Z = 1/1024 (f32), Zh = 1/1024 (f16)
// ---------------------------------------------------------------------------
__global__ __launch_bounds__(256)
void k_fista_init(float* __restrict__ X, float* __restrict__ Z,
                  _Float16* __restrict__ Zh, int n) {
  int i = blockIdx.x * blockDim.x + threadIdx.x;
  if (i < n) {
    const float c = 1.0f / 1024.0f;
    X[i] = c; Z[i] = c; Zh[i] = (_Float16)c;
  }
}

// ---------------------------------------------------------------------------
// FISTA step: G = 2(Z@M - YAt); V = Z - step*G; Xn = proj_simplex(V);
// Zn = Xn + beta*(Xn - X). One wave per 1024-wide row; 32 elems/lane in
// registers; simplex projection via bisection on theta.
// ---------------------------------------------------------------------------
__global__ __launch_bounds__(256)
void k_fista_update(const float* __restrict__ ZM, const float* __restrict__ YAt,
                    float* __restrict__ X, float* __restrict__ Z,
                    _Float16* __restrict__ Zh,
                    const float* __restrict__ stepPtr, float beta) {
  const int row  = blockIdx.x * 8 + (threadIdx.x >> 5);
  const int lane = threadIdx.x & 31;
  const float step = *stepPtr;
  const size_t base = (size_t)row * 1024 + lane;

  float v[32], xo[32];
  float vmax = -1e30f, vsum = 0.f;
  #pragma unroll
  for (int j = 0; j < 32; ++j) {
    const size_t idx = base + (size_t)j * 32;
    const float g   = 2.0f * (ZM[idx] - YAt[idx]);
    const float val = Z[idx] - step * g;
    v[j]  = val;
    xo[j] = X[idx];
    vmax  = fmaxf(vmax, val);
    vsum += val;
  }
  #pragma unroll
  for (int off = 16; off; off >>= 1) {
    vmax  = fmaxf(vmax, __shfl_xor(vmax, off, WAVE));
    vsum += __shfl_xor(vsum, off, WAVE);
  }
  float lo = (vsum - 1.0f) * (1.0f / 1024.0f);   // f(lo) >= 0
  float hi = vmax;                               // f(hi) = -1 < 0
  for (int it = 0; it < 48; ++it) {
    const float th = 0.5f * (lo + hi);
    float s = 0.f;
    #pragma unroll
    for (int j = 0; j < 32; ++j) s += fmaxf(v[j] - th, 0.f);
    #pragma unroll
    for (int off = 16; off; off >>= 1) s += __shfl_xor(s, off, WAVE);
    if (s > 1.0f) lo = th; else hi = th;
  }
  const float th = 0.5f * (lo + hi);
  #pragma unroll
  for (int j = 0; j < 32; ++j) {
    const size_t idx = base + (size_t)j * 32;
    const float xn = fmaxf(v[j] - th, 0.f);
    const float zn = xn + beta * (xn - xo[j]);
    X[idx]  = xn;
    Z[idx]  = zn;
    Zh[idx] = (_Float16)zn;
  }
}

// ---------------------------------------------------------------------------
// Per-row cosine + mean
// ---------------------------------------------------------------------------
__global__ __launch_bounds__(128)
void k_row_cos(const float* __restrict__ pred, const float* __restrict__ y,
               float* __restrict__ cosOut) {
  const int row  = blockIdx.x;
  const int lane = threadIdx.x & 31;
  const int wv   = threadIdx.x >> 5;
  const float* p = pred + (size_t)row * 512;
  const float* q = y    + (size_t)row * 512;
  float dot = 0.f, pp = 0.f;
  for (int c = threadIdx.x; c < 512; c += 128) {
    const float a = p[c];
    dot += a * q[c];
    pp  += a * a;
  }
  #pragma unroll
  for (int off = 16; off; off >>= 1) {
    dot += __shfl_xor(dot, off, WAVE);
    pp  += __shfl_xor(pp,  off, WAVE);
  }
  __shared__ float rd[4], rp[4];
  if (lane == 0) { rd[wv] = dot; rp[wv] = pp; }
  __syncthreads();
  if (threadIdx.x == 0) {
    const float d  = rd[0] + rd[1] + rd[2] + rd[3];
    const float np = sqrtf(rp[0] + rp[1] + rp[2] + rp[3]);
    cosOut[row] = d / (fmaxf(np, 1e-8f) * 1.0f);
  }
}

__global__ __launch_bounds__(1024)
void k_mean_reduce(const float* __restrict__ cosArr, float* __restrict__ out) {
  __shared__ float red[32];
  float s = cosArr[threadIdx.x];
  #pragma unroll
  for (int off = 16; off; off >>= 1) s += __shfl_xor(s, off, WAVE);
  if ((threadIdx.x & 31) == 0) red[threadIdx.x >> 5] = s;
  __syncthreads();
  if (threadIdx.x < 32) {
    float t = red[threadIdx.x];
    #pragma unroll
    for (int off = 16; off; off >>= 1) t += __shfl_xor(t, off, WAVE);
    if (threadIdx.x == 0) *out = t * (1.0f / 1024.0f);
  }
}

// ---------------------------------------------------------------------------
// Host orchestration
// ---------------------------------------------------------------------------
extern "C" void kernel_launch(void* const* d_in, const int* in_sizes, int n_in,
                              void* d_out, int out_size, void* d_ws, size_t ws_size,
                              hipStream_t stream) {
  (void)in_sizes; (void)n_in; (void)out_size; (void)ws_size;
  const float* fea1 = (const float*)d_in[0];   // [1024, 512] f32
  const float* fea2 = (const float*)d_in[1];   // [1024, 512] f32
  float* out = (float*)d_out;

  const int N = 1024, D = 512;
  char* ws = (char*)d_ws;
  size_t off = 0;
  auto carve = [&](size_t bytes) -> void* {
    void* p = ws + off;
    off += (bytes + 255) & ~(size_t)255;
    return p;
  };
  float*    n1f  = (float*)   carve((size_t)N * D * 4);
  float*    n2f  = (float*)   carve((size_t)N * D * 4);
  _Float16* n1h  = (_Float16*)carve((size_t)N * D * 2);
  _Float16* n2h  = (_Float16*)carve((size_t)N * D * 2);
  _Float16* n2Th = (_Float16*)carve((size_t)D * N * 2);
  float*    Mg   = (float*)   carve((size_t)N * N * 4);
  _Float16* Mh   = (_Float16*)carve((size_t)N * N * 2);
  float*    YAt  = (float*)   carve((size_t)N * N * 4);
  float*    X    = (float*)   carve((size_t)N * N * 4);
  float*    Z    = (float*)   carve((size_t)N * N * 4);
  _Float16* Zh   = (_Float16*)carve((size_t)N * N * 2);
  float*    ZM   = (float*)   carve((size_t)N * N * 4);
  _Float16* Xh   = (_Float16*)carve((size_t)N * N * 2);
  float*    pred = (float*)   carve((size_t)N * D * 4);
  float*    cosA = (float*)   carve((size_t)N * 4);
  float*    stepP= (float*)   carve(256);

  // 1) normalize rows
  k_normalize_rows<<<N, 256, 0, stream>>>(fea1, n1f, n1h, D);
  k_normalize_rows<<<N, 256, 0, stream>>>(fea2, n2f, n2h, D);

  // 2) M = A A^T (symmetric Gram), YAt = Y A^T (NT GEMMs, K = 512)
  // block macro-tile = 128x128 -> grid = (M/128)*(N/128)
  k_gemm_nt_f16<<<(N / 128) * (N / 128), 256, 0, stream>>>(n2h, n2h, Mg, N, N, D);
  k_convert_f32_f16<<<(N * N) / 256, 256, 0, stream>>>(Mg, Mh, N * N);
  k_gemm_nt_f16<<<(N / 128) * (N / 128), 256, 0, stream>>>(n1h, n2h, YAt, N, N, D);

  // 3) Lipschitz step via power iteration (f32)
  k_power_iter<<<1, 1024, 0, stream>>>(Mg, stepP);

  // 4) FISTA
  k_fista_init<<<(N * N) / 256, 256, 0, stream>>>(X, Z, Zh, N * N);
  double t = 1.0;
  for (int k = 0; k < 200; ++k) {
    // Z@M = Z@M^T since M symmetric -> NT GEMM, K = 1024
    k_gemm_nt_f16<<<(N / 128) * (N / 128), 256, 0, stream>>>(Zh, Mh, ZM, N, N, N);
    const double tn   = 0.5 * (1.0 + std::sqrt(1.0 + 4.0 * t * t));
    const float  beta = (float)((t - 1.0) / tn);
    k_fista_update<<<N / 8, 256, 0, stream>>>(ZM, YAt, X, Z, Zh, stepP, beta);
    t = tn;
  }

  // 5) pred = X @ nfea2 == NT GEMM of X (f16) against nfea2^T (f16), K = 1024
  k_transpose_f32_to_f16<<<(N * D) / 256, 256, 0, stream>>>(n2f, n2Th, N, D);
  k_convert_f32_f16<<<(N * N) / 256, 256, 0, stream>>>(X, Xh, N * N);
  k_gemm_nt_f16<<<(N / 128) * (D / 128), 256, 0, stream>>>(Xh, n2Th, pred, N, D, N);

  // 6) cosine per row, mean
  k_row_cos<<<N, 128, 0, stream>>>(pred, n1f, cosA);
  k_mean_reduce<<<1, 1024, 0, stream>>>(cosA, out);
}
